// CustomizeLSTMCell_20203526160896
// MI455X (gfx1250) — matrix operand order repaired and checked by reading, not compiled, not run on hardware
//
#include <hip/hip_runtime.h>

// D = A x B + C with A = weight slice (16x32 bf16, register-resident),
// B = activation tile (32x16 bf16 from LDS), C/D = f32 16x16 accumulators.
typedef __attribute__((ext_vector_type(16))) __bf16 v16bf;
typedef __attribute__((ext_vector_type(8)))  float  v8f;

union ABFrag {
  unsigned int u[8];
  uint4        q[2];
  v16bf        v;
};

// Round-to-nearest-even fp32 -> bf16, packed pair (low = first arg).
__device__ __forceinline__ unsigned int pk_bf16(float a, float b) {
  unsigned int ua = __float_as_uint(a);
  unsigned int ub = __float_as_uint(b);
  ua += 0x7FFFu + ((ua >> 16) & 1u);
  ub += 0x7FFFu + ((ub >> 16) & 1u);
  return (ua >> 16) | (ub & 0xFFFF0000u);
}

// H = S = 128 fixed by the reference; Bcols = batch width (131072).
// One workgroup: 8 waves, wave w computes H-rows [16w,16w+16) for a 32-column
// batch tile. Activations staged once per tile into LDS as bf16 [n][k].
__global__ __launch_bounds__(256)
void lstm_fused_wmma(const float* __restrict__ s_in,  const float* __restrict__ s_out,
                     const float* __restrict__ h_in,  const float* __restrict__ h_out,
                     const float* __restrict__ last_c,
                     const float* __restrict__ w_in,  const float* __restrict__ w_out,
                     const float* __restrict__ u_in,  const float* __restrict__ u_out,
                     float* __restrict__ out, int Bcols, int ntiles) {
  constexpr int P = 520;                      // LDS pitch in bf16 elems (bank spread)
  __shared__ unsigned short xtile[32 * P];    // 32 cols x 512 k, bf16 (~33 KB)

  const int t     = threadIdx.x;
  const int wave  = t >> 5;
  const int lane  = t & 31;
  const int lhalf = lane >> 4;                // 0: K/M low half, 1: high half
  const int l15   = lane & 15;

  // ---- Preload this wave's 16x512 bf16 slice of Wcat into registers ----
  // A 16x32 bf16 layout: lane holds row M=l15; VGPR v holds
  // K = (v>>2)*16 + lhalf*8 + (v&3)*2 (and +1 in the high 16 bits).
  const float* wptr[4] = { w_in, w_out, u_in, u_out };
  const int row = wave * 16 + l15;
  const int h8  = lhalf * 8;
  ABFrag a[16];
#pragma unroll
  for (int ks = 0; ks < 16; ++ks) {
    const float* wp = wptr[ks >> 2] + (size_t)row * 128 + (ks & 3) * 32;
    float4 q0 = *(const float4*)(wp + h8);
    float4 q1 = *(const float4*)(wp + h8 + 4);
    float4 q2 = *(const float4*)(wp + h8 + 16);
    float4 q3 = *(const float4*)(wp + h8 + 20);
    a[ks].u[0] = pk_bf16(q0.x, q0.y); a[ks].u[1] = pk_bf16(q0.z, q0.w);
    a[ks].u[2] = pk_bf16(q1.x, q1.y); a[ks].u[3] = pk_bf16(q1.z, q1.w);
    a[ks].u[4] = pk_bf16(q2.x, q2.y); a[ks].u[5] = pk_bf16(q2.z, q2.w);
    a[ks].u[6] = pk_bf16(q3.x, q3.y); a[ks].u[7] = pk_bf16(q3.z, q3.w);
  }

  const float* act[4] = { s_in, s_out, h_in, h_out };
  const size_t HB = (size_t)128 * (size_t)Bcols;

  for (int tile = blockIdx.x; tile < ntiles; tile += gridDim.x) {
    const int col0 = tile * 32;

    // ---- Cooperative staging: X[512 x 32] fp32 -> LDS bf16 [n][k] ----
    // Thread handles k-rows (k,k+1) x 4 consecutive columns; 128B-contiguous
    // global reads per 8-lane group, packed-pair 4B LDS stores.
    {
      const int n2 = (t & 7) * 4;
      const int kb = (t >> 3) * 2;
#pragma unroll
      for (int pass = 0; pass < 8; ++pass) {
        const int k = pass * 64 + kb;                         // even, same 128-blk as k+1
        const float* ap = act[k >> 7] + (size_t)(k & 127) * Bcols + col0 + n2;
        float4 r0 = *(const float4*)(ap);
        float4 r1 = *(const float4*)(ap + Bcols);
        *(unsigned int*)&xtile[(n2 + 0) * P + k] = pk_bf16(r0.x, r1.x);
        *(unsigned int*)&xtile[(n2 + 1) * P + k] = pk_bf16(r0.y, r1.y);
        *(unsigned int*)&xtile[(n2 + 2) * P + k] = pk_bf16(r0.z, r1.z);
        *(unsigned int*)&xtile[(n2 + 3) * P + k] = pk_bf16(r0.w, r1.w);
      }
    }
    __syncthreads();

    // ---- 16 K-steps x 2 N-subtiles of v_wmma_f32_16x16x32_bf16 ----
    // B 32x16 layout: lane -> column N=l15, elems 0-7 hold K=lhalf*8+0..7,
    // elems 8-15 hold K=16+lhalf*8+0..7 => 2x b128 from LDS.
    v8f c0 = {0.f, 0.f, 0.f, 0.f, 0.f, 0.f, 0.f, 0.f};
    v8f c1 = c0;
#pragma unroll
    for (int ks = 0; ks < 16; ++ks) {
      const int kb = ks * 32 + lhalf * 8;
      ABFrag b0, b1;
      const uint4* p0 = (const uint4*)&xtile[l15 * P + kb];
      const uint4* p1 = (const uint4*)&xtile[(l15 + 16) * P + kb];
      b0.q[0] = p0[0]; b0.q[1] = p0[2];
      b1.q[0] = p1[0]; b1.q[1] = p1[2];
      c0 = __builtin_amdgcn_wmma_f32_16x16x32_bf16(false, a[ks].v, false, b0.v,
                                                   (short)0, c0, false, false);
      c1 = __builtin_amdgcn_wmma_f32_16x16x32_bf16(false, a[ks].v, false, b1.v,
                                                   (short)0, c1, false, false);
    }

    // ---- Epilogue: g = sigmoid(pre); cell = g*c_prev + g*g; hidden = g*cell ----
    // C/D layout: VGPR r -> M = wave*16 + lhalf*8 + r, N = col0 + l15 (+16).
    {
      const int ncol  = col0 + l15;
      const int mbase = wave * 16 + lhalf * 8;
#pragma unroll
      for (int r = 0; r < 8; ++r) {
        {
          const size_t off = (size_t)(mbase + r) * Bcols + ncol;
          float g    = __builtin_amdgcn_rcpf(1.0f + __expf(-c0[r]));
          float cell = g * last_c[off] + g * g;
          out[off]      = cell;
          out[HB + off] = g * cell;
        }
        {
          const size_t off = (size_t)(mbase + r) * Bcols + ncol + 16;
          float g    = __builtin_amdgcn_rcpf(1.0f + __expf(-c1[r]));
          float cell = g * last_c[off] + g * g;
          out[off]      = cell;
          out[HB + off] = g * cell;
        }
      }
    }
    __syncthreads();   // LDS fully consumed before next tile's staging
  }
}

extern "C" void kernel_launch(void* const* d_in, const int* in_sizes, int n_in,
                              void* d_out, int out_size, void* d_ws, size_t ws_size,
                              hipStream_t stream) {
  (void)n_in; (void)out_size; (void)d_ws; (void)ws_size;
  const float* s_in   = (const float*)d_in[0];
  const float* s_out  = (const float*)d_in[1];
  const float* h_in   = (const float*)d_in[2];
  const float* h_out  = (const float*)d_in[3];
  const float* last_c = (const float*)d_in[4];
  const float* w_in   = (const float*)d_in[5];
  const float* w_out  = (const float*)d_in[6];
  const float* u_in   = (const float*)d_in[7];
  const float* u_out  = (const float*)d_in[8];
  float* out = (float*)d_out;

  const int Bcols  = in_sizes[0] / 128;   // S = 128
  const int ntiles = Bcols / 32;
  if (ntiles <= 0) return;
  // Grid-stride: enough blocks to saturate the device, few enough that the
  // register-resident weight preload is amortized over >=2 tiles per block.
  int grid = ntiles < 2048 ? ntiles : 2048;

  lstm_fused_wmma<<<grid, 256, 0, stream>>>(s_in, s_out, h_in, h_out, last_c,
                                            w_in, w_out, u_in, u_out,
                                            out, Bcols, ntiles);
}
